// GINNet_2336462209633
// MI455X (gfx1250) — matrix operand order repaired and checked by reading, not compiled, not run on hardware
//
#include <hip/hip_runtime.h>
#include <hip/hip_bf16.h>

// GIN forward for MI455X (gfx1250, wave32, WMMA).
// N=50000 nodes, D=H=128, O=64, L=3, E=800000 edges.
//  - scatter: one wave per edge, 4 f32 atomics/lane into L2-resident agg table
//  - GEMMs:   v_wmma_f32_16x16x32_bf16, one wave = 16xNC output stripe,
//             A fragment converted f32->bf16 in-register (fused x+agg for GEMM1)
//  - epilogue: bias + relu + deterministic hash dropout (p=0.1)

#define DF 128  // K dimension of every GEMM (feature width)

typedef __attribute__((ext_vector_type(16))) __bf16 bf16x16;
typedef __attribute__((ext_vector_type(8)))  float  f32x8;

__device__ __forceinline__ unsigned mix32(unsigned x) {
  x ^= x >> 16; x *= 0x7feb352dU;
  x ^= x >> 15; x *= 0x846ca68bU;
  x ^= x >> 16;
  return x;
}

// ---------------- zero the aggregation buffer ----------------
__global__ __launch_bounds__(256)
void zero_f32(float* __restrict__ p, long n4) {
  long i      = (long)blockIdx.x * blockDim.x + threadIdx.x;
  long stride = (long)gridDim.x * blockDim.x;
  float4 z = make_float4(0.f, 0.f, 0.f, 0.f);
  for (; i < n4; i += stride) ((float4*)p)[i] = z;
}

// ---------------- edge scatter: agg[dst] += x[src] ----------------
// One wave32 per edge; each lane moves 4 floats (4 x global_atomic_add_f32).
__global__ __launch_bounds__(256)
void scatter_edges(const float* __restrict__ x,
                   const long long* __restrict__ src,
                   const long long* __restrict__ dst,
                   float* __restrict__ agg, int E) {
  int wave = (int)((blockIdx.x * (unsigned)blockDim.x + threadIdx.x) >> 5);
  int lane = threadIdx.x & 31;
  if (wave >= E) return;
  long long s = src[wave];
  long long d = dst[wave];
  float4 v = *(const float4*)(x + (size_t)s * DF + lane * 4);
  float* p = agg + (size_t)d * DF + lane * 4;
  __hip_atomic_fetch_add(p + 0, v.x, __ATOMIC_RELAXED, __HIP_MEMORY_SCOPE_AGENT);
  __hip_atomic_fetch_add(p + 1, v.y, __ATOMIC_RELAXED, __HIP_MEMORY_SCOPE_AGENT);
  __hip_atomic_fetch_add(p + 2, v.z, __ATOMIC_RELAXED, __HIP_MEMORY_SCOPE_AGENT);
  __hip_atomic_fetch_add(p + 3, v.w, __ATOMIC_RELAXED, __HIP_MEMORY_SCOPE_AGENT);
}

// ---------------- weight prep: W[K][N] f32 -> Wt[N][K] bf16 ----------------
__global__ __launch_bounds__(256)
void prep_wt(const float* __restrict__ W, __bf16* __restrict__ Wt, int K, int N) {
  int idx = blockIdx.x * blockDim.x + threadIdx.x;
  if (idx >= K * N) return;
  int n = idx / K;
  int k = idx - n * K;
  Wt[idx] = (__bf16)W[k * N + n];
}

// ---------------- WMMA GEMM: C[M][NC] = post(A (+A2) @ W + bias) ------------
// One wave computes a 16 x NC stripe. K fixed at 128 (4 k-tiles of 32).
// Wt layout: [NC][128] bf16 (row = output column), so a lane's B-fragment
// (N = lane&15, K = k0+8*(lane>>4)+i and that +16, i<8) is two 16-byte runs.
template <int NC, bool FUSE, bool RELU, bool DROP>
__global__ __launch_bounds__(256)
void gemm_wmma(const float* __restrict__ A, const float* __restrict__ A2,
               const __bf16* __restrict__ Wt, const float* __restrict__ bias,
               float* __restrict__ C, int M, unsigned seed) {
  constexpr int NT = NC / 16;
  int gw   = (int)((blockIdx.x * (unsigned)blockDim.x + threadIdx.x) >> 5);
  int lane = threadIdx.x & 31;
  int tm   = gw * 16;                 // M-tile base (M is a multiple of 16 here)
  if (tm >= M) return;                // wave-uniform: EXEC stays all-ones below
  int sub  = lane & 15;
  int half = lane >> 4;

  f32x8 acc[NT];
  f32x8 zero = {};
#pragma unroll
  for (int t = 0; t < NT; ++t) acc[t] = zero;

  const float* arow  = A + (size_t)(tm + sub) * DF;
  const float* arow2 = FUSE ? (A2 + (size_t)(tm + sub) * DF) : nullptr;

#pragma unroll
  for (int k0 = 0; k0 < DF; k0 += 32) {
    // A fragment (16-bit A 16x32 layout): lane(row=sub), elems 0..7 = K
    // k0+8*half+i, elems 8..15 = K k0+16+8*half+i.
    bf16x16 a;
    int kb = k0 + half * 8;
#pragma unroll
    for (int i = 0; i < 8; ++i) {
      float v0 = arow[kb + i];
      float v1 = arow[kb + 16 + i];
      if (FUSE) {
        v0 += arow2[kb + i];
        v1 += arow2[kb + 16 + i];
      }
      a[i]     = (__bf16)v0;
      a[i + 8] = (__bf16)v1;
    }
    // Reuse the A fragment across all N-tiles of this stripe.
#pragma unroll
    for (int t = 0; t < NT; ++t) {
      const __bf16* wp = Wt + (size_t)(t * 16 + sub) * DF + k0 + half * 8;
      typedef __attribute__((ext_vector_type(8))) __bf16 bf16x8_t;
      bf16x8_t blo = *(const bf16x8_t*)wp;
      bf16x8_t bhi = *(const bf16x8_t*)(wp + 16);
      bf16x16 b = __builtin_shufflevector(blo, bhi, 0, 1, 2, 3, 4, 5, 6, 7, 8, 9, 10, 11, 12, 13, 14, 15);
      acc[t] = __builtin_amdgcn_wmma_f32_16x16x32_bf16(
          false, a, false, b, (short)0, acc[t], false, false);
    }
  }

  // C layout: VGPR i -> row tm + 8*half + i, col = t*16 + sub.
  int mb = tm + half * 8;
  const float    inv_keep = 1.0f / 0.9f;
  const unsigned thr      = 429496730u;  // 0.1 * 2^32
#pragma unroll
  for (int t = 0; t < NT; ++t) {
    int   n  = t * 16 + sub;
    float bv = bias[n];
#pragma unroll
    for (int i = 0; i < 8; ++i) {
      float v = acc[t][i] + bv;
      if (RELU) v = fmaxf(v, 0.0f);
      if (DROP) {
        unsigned h = mix32(seed * 0x9E3779B9u + (unsigned)((mb + i) * NC + n));
        v = (h >= thr) ? v * inv_keep : 0.0f;
      }
      C[(size_t)(mb + i) * NC + n] = v;
    }
  }
}

extern "C" void kernel_launch(void* const* d_in, const int* in_sizes, int n_in,
                              void* d_out, int out_size, void* d_ws, size_t ws_size,
                              hipStream_t stream) {
  const float*     x    = (const float*)d_in[0];
  const long long* ei   = (const long long*)d_in[1];  // int64 edge_index [2,E]
  const float*     Ws1  = (const float*)d_in[2];      // [3,128,128]
  const float*     bs1  = (const float*)d_in[3];      // [3,128]
  const float*     Ws2  = (const float*)d_in[4];      // [3,128,128]
  const float*     bs2  = (const float*)d_in[5];      // [3,128]
  const float*     Wlin = (const float*)d_in[6];      // [128,64]
  const float*     blin = (const float*)d_in[7];      // [64]

  const int N = in_sizes[0] / DF;   // 50000 (multiple of 16)
  const int E = in_sizes[1] / 2;    // 800000
  const long long* src = ei;
  const long long* dst = ei + E;

  // Workspace carve-up.
  size_t  nd  = (size_t)N * DF;
  float*  agg = (float*)d_ws;
  float*  tmp = agg + nd;
  float*  hA  = tmp + nd;
  float*  hB  = hA + nd;
  __bf16* wt1 = (__bf16*)(hB + nd);        // 3 x [128][128]
  __bf16* wt2 = wt1 + (size_t)3 * DF * DF; // 3 x [128][128]
  __bf16* wtl = wt2 + (size_t)3 * DF * DF; // [64][128]

  // Pre-transpose + bf16-convert weights (tiny; runs once per launch).
  for (int l = 0; l < 3; ++l) {
    prep_wt<<<(DF * DF + 255) / 256, 256, 0, stream>>>(
        Ws1 + (size_t)l * DF * DF, wt1 + (size_t)l * DF * DF, DF, DF);
    prep_wt<<<(DF * DF + 255) / 256, 256, 0, stream>>>(
        Ws2 + (size_t)l * DF * DF, wt2 + (size_t)l * DF * DF, DF, DF);
  }
  prep_wt<<<(DF * 64 + 255) / 256, 256, 0, stream>>>(Wlin, wtl, DF, 64);

  const float* cur    = x;
  float*       buf[2] = {hA, hB};
  const int    gemm_blocks = ((N / 16) + 7) / 8;  // 8 waves per 256-thread block

  for (int l = 0; l < 3; ++l) {
    zero_f32<<<1024, 256, 0, stream>>>(agg, (long)(nd / 4));
    scatter_edges<<<(E + 7) / 8, 256, 0, stream>>>(cur, src, dst, agg, E);
    // h = relu((x + agg) @ W1 + b1)      (x+agg fused into A-fragment load)
    gemm_wmma<128, true, true, false><<<gemm_blocks, 256, 0, stream>>>(
        cur, agg, wt1 + (size_t)l * DF * DF, bs1 + l * DF, tmp, N, 0u);
    // h = dropout(relu(h @ W2 + b2))
    gemm_wmma<128, false, true, true><<<gemm_blocks, 256, 0, stream>>>(
        tmp, nullptr, wt2 + (size_t)l * DF * DF, bs2 + l * DF, buf[l & 1], N,
        0xC0FFEEu + (unsigned)l);
    cur = buf[l & 1];
  }

  // Final linear: out[N][64] = h @ Wlin + blin
  gemm_wmma<64, false, false, false><<<gemm_blocks, 256, 0, stream>>>(
      cur, nullptr, wtl, blin, (float*)d_out, N, 0u);
}
